// SelfAttention_4028679324048
// MI455X (gfx1250) — matrix operand order repaired and checked by reading, not compiled, not run on hardware
//
#include <hip/hip_runtime.h>
#include <hip/hip_bf16.h>
#include <math.h>
#include <stdint.h>

#define B_ 8
#define S_ 2048
#define D_ 512
#define U_ 512

#define MT 128   // M tile per workgroup
#define NT 128   // N tile per workgroup
#define KT 32    // K step (one bf16 WMMA depth)
#define PAD 40   // LDS row stride in bf16 elements (80B): 16B-aligned, staggers banks

typedef __bf16 bf16_t;
typedef __attribute__((ext_vector_type(16))) __bf16 v16bf;
typedef __attribute__((ext_vector_type(8)))  __bf16 v8bf;
typedef __attribute__((ext_vector_type(8)))  float  v8f;

union FragAB { v16bf v; v8bf h[2]; };

struct KTile {
    bf16_t A[MT * PAD];    // A[m][k], row-major, padded
    bf16_t Bt[NT * PAD];   // Bt[n][k] (K contiguous per column), padded
};

// ---------------------------------------------------------------------------
// f32 -> bf16 cast (grid-stride)
// ---------------------------------------------------------------------------
__global__ __launch_bounds__(256)
void cast_f32_bf16(const float* __restrict__ in, bf16_t* __restrict__ out, long n) {
    long i = (long)blockIdx.x * blockDim.x + threadIdx.x;
    long stride = (long)gridDim.x * blockDim.x;
    for (; i < n; i += stride) out[i] = (bf16_t)in[i];
}

// ---------------------------------------------------------------------------
// One-shot tiled transposes (64x64 tiles through LDS; wide loads + wide stores)
// ---------------------------------------------------------------------------
__global__ __launch_bounds__(256)
void transpose_cast_f32_bf16(const float* __restrict__ in, bf16_t* __restrict__ out,
                             int R, int C) {
    // in: R x C f32 (row-major) -> out: C x R bf16 (row-major)
    __shared__ bf16_t t[64][72];   // 144B rows: 16B-aligned, staggered banks
    const int tid = threadIdx.x;
    const int r0 = blockIdx.x * 64, c0 = blockIdx.y * 64;

    int r = tid >> 2, cs = (tid & 3) * 16;
    const float* src = in + (size_t)(r0 + r) * C + c0 + cs;
    v8bf a, b;
#pragma unroll
    for (int j = 0; j < 8; ++j) a[j] = (bf16_t)src[j];
#pragma unroll
    for (int j = 0; j < 8; ++j) b[j] = (bf16_t)src[8 + j];
    *(v8bf*)&t[r][cs]     = a;
    *(v8bf*)&t[r][cs + 8] = b;
    __syncthreads();

    int c = tid >> 2, rs = (tid & 3) * 16;
    v8bf o0, o1;
#pragma unroll
    for (int j = 0; j < 8; ++j) o0[j] = t[rs + j][c];
#pragma unroll
    for (int j = 0; j < 8; ++j) o1[j] = t[rs + 8 + j][c];
    bf16_t* dst = out + (size_t)(c0 + c) * R + r0 + rs;
    *(v8bf*)dst       = o0;
    *(v8bf*)(dst + 8) = o1;
}

__global__ __launch_bounds__(256)
void transpose_bf16(const bf16_t* __restrict__ in, bf16_t* __restrict__ out,
                    int R, int C) {
    // per-batch: in + z*R*C (R x C) -> out + z*C*R (C x R)
    __shared__ bf16_t t[64][72];
    const int tid = threadIdx.x;
    const int r0 = blockIdx.x * 64, c0 = blockIdx.y * 64;
    const bf16_t* bin  = in  + (size_t)blockIdx.z * R * C;
    bf16_t*       bout = out + (size_t)blockIdx.z * C * R;

    int r = tid >> 2, cs = (tid & 3) * 16;
    const bf16_t* src = bin + (size_t)(r0 + r) * C + c0 + cs;
    *(v8bf*)&t[r][cs]     = *(const v8bf*)src;
    *(v8bf*)&t[r][cs + 8] = *(const v8bf*)(src + 8);
    __syncthreads();

    int c = tid >> 2, rs = (tid & 3) * 16;
    v8bf o0, o1;
#pragma unroll
    for (int j = 0; j < 8; ++j) o0[j] = t[rs + j][c];
#pragma unroll
    for (int j = 0; j < 8; ++j) o1[j] = t[rs + 8 + j][c];
    bf16_t* dst = bout + (size_t)(c0 + c) * R + r0 + rs;
    *(v8bf*)dst       = o0;
    *(v8bf*)(dst + 8) = o1;
}

// ---------------------------------------------------------------------------
// Async staging: cache -> LDS direct (no VGPR round trip), tracked by ASYNCcnt.
// Each thread stages 32B of a 128x32 bf16 tile as two async b128 copies.
// INST_OFFSET is added to BOTH the LDS and global addresses (ISA 15.18.3).
// ---------------------------------------------------------------------------
__device__ __forceinline__
void stage_rows_async(bf16_t* lds, const bf16_t* __restrict__ g, int ldg, int tid) {
    int row  = tid >> 1;
    int half = (tid & 1) * 16;
    const bf16_t* src = g + (size_t)row * ldg + half;
    bf16_t* dst = lds + row * PAD + half;
    unsigned lds_off = (unsigned)(uintptr_t)dst;   // low 32 bits of flat = LDS offset
    asm volatile("global_load_async_to_lds_b128 %0, %1, off"
                 :: "v"(lds_off), "v"(src) : "memory");
    asm volatile("global_load_async_to_lds_b128 %0, %1, off offset:16"
                 :: "v"(lds_off), "v"(src) : "memory");
}

__device__ __forceinline__ void wait_async_all() {
    asm volatile("s_wait_asynccnt 0x0" ::: "memory");
}

// ---------------------------------------------------------------------------
// One K-step of WMMA for a wave's 32x64 sub-tile: 2(M) x 4(N) fragments
// A layout per ISA: lanes 0-15 row m, K[0..7]+[16..23]; lanes 16-31 K[8..15]+[24..31]
// B layout: lanes 0-15 col n, K[0..7]+[16..23]; lanes 16-31 K[8..15]+[24..31]
// ---------------------------------------------------------------------------
__device__ __forceinline__
void wmma_step(const bf16_t* ldsA, const bf16_t* ldsBt,
               int lane, int wm0, int wn0, v8f acc[2][4]) {
    const int lrow   = lane & 15;
    const int khalfA = (lane >> 4) * 8;
    const int khalfB = (lane >> 4) * 8;

    FragAB a[2];
#pragma unroll
    for (int mi = 0; mi < 2; ++mi) {
        const bf16_t* pa = ldsA + (wm0 + mi * 16 + lrow) * PAD + khalfA;
        a[mi].h[0] = *(const v8bf*)pa;         // K 0..7  (or 8..15)
        a[mi].h[1] = *(const v8bf*)(pa + 16);  // K 16..23 (or 24..31)
    }
#pragma unroll
    for (int ni = 0; ni < 4; ++ni) {
        FragAB b;
        const bf16_t* pb = ldsBt + (wn0 + ni * 16 + lrow) * PAD + khalfB;
        b.h[0] = *(const v8bf*)pb;             // K 0..7  (or 8..15)
        b.h[1] = *(const v8bf*)(pb + 16);      // K 16..23 (or 24..31)
#pragma unroll
        for (int mi = 0; mi < 2; ++mi) {
            acc[mi][ni] = __builtin_amdgcn_wmma_f32_16x16x32_bf16(
                false, a[mi].v, false, b.v, (short)0, acc[mi][ni], false, false);
        }
    }
}

__device__ __forceinline__ void zero_acc(v8f acc[2][4]) {
#pragma unroll
    for (int mi = 0; mi < 2; ++mi)
#pragma unroll
        for (int ni = 0; ni < 4; ++ni)
#pragma unroll
            for (int j = 0; j < 8; ++j) acc[mi][ni][j] = 0.0f;
}

// ---------------------------------------------------------------------------
// Kernel B: QKV projection  [B*S, D] @ Wt[U, D]^T -> bf16 Q/K/V  (z selects W)
// ---------------------------------------------------------------------------
__global__ __launch_bounds__(256)
void qkv_gemm(const bf16_t* __restrict__ Xb,
              const bf16_t* __restrict__ Wtq, const bf16_t* __restrict__ Wtk,
              const bf16_t* __restrict__ Wtv,
              bf16_t* __restrict__ Qb, bf16_t* __restrict__ Kb, bf16_t* __restrict__ Vb) {
    const bf16_t* Wt = (blockIdx.z == 0) ? Wtq : (blockIdx.z == 1) ? Wtk : Wtv;
    bf16_t*       O  = (blockIdx.z == 0) ? Qb  : (blockIdx.z == 1) ? Kb  : Vb;

    const int m0 = blockIdx.x * MT;   // over B*S
    const int n0 = blockIdx.y * NT;   // over U
    __shared__ KTile tiles[2];

    const int tid  = threadIdx.x;
    const int lane = tid & 31;
    const int wave = tid >> 5;
    const int wm0  = (wave & 3) * 32;
    const int wn0  = (wave >> 2) * 64;

    v8f acc[2][4];
    zero_acc(acc);

    stage_rows_async(tiles[0].A,  Xb + (size_t)m0 * D_, D_, tid);
    stage_rows_async(tiles[0].Bt, Wt + (size_t)n0 * D_, D_, tid);

    int cur = 0;
    for (int k0 = 0; k0 < D_; k0 += KT) {
        wait_async_all();
        __syncthreads();
        if (k0 + KT < D_) {
            stage_rows_async(tiles[cur ^ 1].A,  Xb + (size_t)m0 * D_ + k0 + KT, D_, tid);
            stage_rows_async(tiles[cur ^ 1].Bt, Wt + (size_t)n0 * D_ + k0 + KT, D_, tid);
            if (k0 + 2 * KT < D_)
                __builtin_prefetch(Xb + (size_t)(m0 + (tid >> 1)) * D_ + k0 + 2 * KT, 0, 3);
        }
        wmma_step(tiles[cur].A, tiles[cur].Bt, lane, wm0, wn0, acc);
        cur ^= 1;
    }

    const int lrow = lane & 15;
    const int lhi  = (lane >> 4) * 8;
#pragma unroll
    for (int mi = 0; mi < 2; ++mi)
#pragma unroll
        for (int ni = 0; ni < 4; ++ni)
#pragma unroll
            for (int r = 0; r < 8; ++r) {
                int row = m0 + wm0 + mi * 16 + lhi + r;
                int col = n0 + wn0 + ni * 16 + lrow;
                O[(size_t)row * U_ + col] = (bf16_t)acc[mi][ni][r];
            }
}

// ---------------------------------------------------------------------------
// Kernel C: scores = Q @ K^T * scale -> f32 attn region (pre-softmax)
// ---------------------------------------------------------------------------
__global__ __launch_bounds__(256)
void scores_gemm(const bf16_t* __restrict__ Qb, const bf16_t* __restrict__ Kb,
                 float* __restrict__ attn, float scale) {
    const int b  = blockIdx.z;
    const int m0 = blockIdx.x * MT;   // over S
    const int n0 = blockIdx.y * NT;   // over S
    const bf16_t* Qp = Qb + (size_t)b * S_ * U_;
    const bf16_t* Kp = Kb + (size_t)b * S_ * U_;

    __shared__ KTile tiles[2];

    const int tid  = threadIdx.x;
    const int lane = tid & 31;
    const int wave = tid >> 5;
    const int wm0  = (wave & 3) * 32;
    const int wn0  = (wave >> 2) * 64;

    v8f acc[2][4];
    zero_acc(acc);

    stage_rows_async(tiles[0].A,  Qp + (size_t)m0 * U_, U_, tid);
    stage_rows_async(tiles[0].Bt, Kp + (size_t)n0 * U_, U_, tid);

    int cur = 0;
    for (int k0 = 0; k0 < U_; k0 += KT) {
        wait_async_all();
        __syncthreads();
        if (k0 + KT < U_) {
            stage_rows_async(tiles[cur ^ 1].A,  Qp + (size_t)m0 * U_ + k0 + KT, U_, tid);
            stage_rows_async(tiles[cur ^ 1].Bt, Kp + (size_t)n0 * U_ + k0 + KT, U_, tid);
            if (k0 + 2 * KT < U_)
                __builtin_prefetch(Qp + (size_t)(m0 + (tid >> 1)) * U_ + k0 + 2 * KT, 0, 3);
        }
        wmma_step(tiles[cur].A, tiles[cur].Bt, lane, wm0, wn0, acc);
        cur ^= 1;
    }

    const int lrow = lane & 15;
    const int lhi  = (lane >> 4) * 8;
    float* out = attn + (size_t)b * S_ * S_;
#pragma unroll
    for (int mi = 0; mi < 2; ++mi)
#pragma unroll
        for (int ni = 0; ni < 4; ++ni)
#pragma unroll
            for (int r = 0; r < 8; ++r) {
                int row = m0 + wm0 + mi * 16 + lhi + r;
                int col = n0 + wn0 + ni * 16 + lrow;
                out[(size_t)row * S_ + col] = acc[mi][ni][r] * scale;
            }
}

// ---------------------------------------------------------------------------
// Kernel D: row softmax; writes normalized f32 in place AND a bf16 copy
// ---------------------------------------------------------------------------
__global__ __launch_bounds__(256)
void softmax_rows(float* __restrict__ attn, bf16_t* __restrict__ attn_bf) {
    const size_t row = blockIdx.x;
    float*  p  = attn    + row * (size_t)S_;
    bf16_t* pb = attn_bf + row * (size_t)S_;
    const int tid = threadIdx.x;

    float vals[S_ / 256];
    float m = -INFINITY;
#pragma unroll
    for (int j = 0; j < S_ / 256; ++j) {
        vals[j] = p[tid + j * 256];
        m = fmaxf(m, vals[j]);
    }
    __shared__ float red[256];
    red[tid] = m;
    __syncthreads();
    for (int s = 128; s > 0; s >>= 1) {
        if (tid < s) red[tid] = fmaxf(red[tid], red[tid + s]);
        __syncthreads();
    }
    m = red[0];
    __syncthreads();

    float sum = 0.0f;
#pragma unroll
    for (int j = 0; j < S_ / 256; ++j) {
        vals[j] = __expf(vals[j] - m);
        sum += vals[j];
    }
    red[tid] = sum;
    __syncthreads();
    for (int s = 128; s > 0; s >>= 1) {
        if (tid < s) red[tid] += red[tid + s];
        __syncthreads();
    }
    const float inv = 1.0f / red[0];
#pragma unroll
    for (int j = 0; j < S_ / 256; ++j) {
        float v = vals[j] * inv;
        p[tid + j * 256]  = v;
        pb[tid + j * 256] = (bf16_t)v;
    }
}

// ---------------------------------------------------------------------------
// Kernel E: context = attn_bf16 @ V  (B from pre-transposed Vt[U][S] per batch)
// ---------------------------------------------------------------------------
__global__ __launch_bounds__(256)
void context_gemm(const bf16_t* __restrict__ attn_bf, const bf16_t* __restrict__ Vt,
                  float* __restrict__ ctx) {
    const int b  = blockIdx.z;
    const int m0 = blockIdx.x * MT;   // over S
    const int n0 = blockIdx.y * NT;   // over U
    const bf16_t* Ap  = attn_bf + (size_t)b * S_ * S_;
    const bf16_t* Vtp = Vt      + (size_t)b * U_ * S_;

    __shared__ KTile tiles[2];

    const int tid  = threadIdx.x;
    const int lane = tid & 31;
    const int wave = tid >> 5;
    const int wm0  = (wave & 3) * 32;
    const int wn0  = (wave >> 2) * 64;

    v8f acc[2][4];
    zero_acc(acc);

    stage_rows_async(tiles[0].A,  Ap  + (size_t)m0 * S_, S_, tid);
    stage_rows_async(tiles[0].Bt, Vtp + (size_t)n0 * S_, S_, tid);

    int cur = 0;
    for (int k0 = 0; k0 < S_; k0 += KT) {
        wait_async_all();
        __syncthreads();
        if (k0 + KT < S_) {
            stage_rows_async(tiles[cur ^ 1].A,  Ap  + (size_t)m0 * S_ + k0 + KT, S_, tid);
            stage_rows_async(tiles[cur ^ 1].Bt, Vtp + (size_t)n0 * S_ + k0 + KT, S_, tid);
            if (k0 + 2 * KT < S_)
                __builtin_prefetch(Ap + (size_t)(m0 + (tid >> 1)) * S_ + k0 + 2 * KT, 0, 3);
        }
        wmma_step(tiles[cur].A, tiles[cur].Bt, lane, wm0, wn0, acc);
        cur ^= 1;
    }

    const int lrow = lane & 15;
    const int lhi  = (lane >> 4) * 8;
    float* out = ctx + (size_t)b * S_ * U_;
#pragma unroll
    for (int mi = 0; mi < 2; ++mi)
#pragma unroll
        for (int ni = 0; ni < 4; ++ni)
#pragma unroll
            for (int r = 0; r < 8; ++r) {
                int row = m0 + wm0 + mi * 16 + lhi + r;
                int col = n0 + wn0 + ni * 16 + lrow;
                out[(size_t)row * U_ + col] = acc[mi][ni][r];
            }
}

// ---------------------------------------------------------------------------
// Host launcher
// ---------------------------------------------------------------------------
extern "C" void kernel_launch(void* const* d_in, const int* in_sizes, int n_in,
                              void* d_out, int out_size, void* d_ws, size_t ws_size,
                              hipStream_t stream) {
    (void)in_sizes; (void)n_in; (void)out_size; (void)ws_size;

    const float* X  = (const float*)d_in[0];
    const float* Wq = (const float*)d_in[1];
    const float* Wk = (const float*)d_in[2];
    const float* Wv = (const float*)d_in[3];

    const size_t nX = (size_t)B_ * S_ * D_;   // 8,388,608
    const size_t nW = (size_t)D_ * U_;        //   262,144
    const size_t nQ = (size_t)B_ * S_ * U_;   // 8,388,608
    const size_t nA = (size_t)B_ * S_ * S_;   // 33,554,432

    char* ws = (char*)d_ws;
    bf16_t* Xb   = (bf16_t*)ws;  ws += nX * sizeof(bf16_t);
    bf16_t* Wtq  = (bf16_t*)ws;  ws += nW * sizeof(bf16_t);
    bf16_t* Wtk  = (bf16_t*)ws;  ws += nW * sizeof(bf16_t);
    bf16_t* Wtv  = (bf16_t*)ws;  ws += nW * sizeof(bf16_t);
    bf16_t* Qb   = (bf16_t*)ws;  ws += nQ * sizeof(bf16_t);
    bf16_t* Kb   = (bf16_t*)ws;  ws += nQ * sizeof(bf16_t);
    bf16_t* Vb   = (bf16_t*)ws;  ws += nQ * sizeof(bf16_t);
    bf16_t* Vtb  = (bf16_t*)ws;  ws += nQ * sizeof(bf16_t);
    bf16_t* Abf  = (bf16_t*)ws;  ws += nA * sizeof(bf16_t);
    // total workspace ~153 MB

    float* ctx  = (float*)d_out;          // [B,S,U]
    float* attn = ctx + nQ;               // [B,S,S]

    dim3 blk(256);
    cast_f32_bf16<<<dim3(4096), blk, 0, stream>>>(X, Xb, (long)nX);
    transpose_cast_f32_bf16<<<dim3(D_ / 64, U_ / 64), blk, 0, stream>>>(Wq, Wtq, D_, U_);
    transpose_cast_f32_bf16<<<dim3(D_ / 64, U_ / 64), blk, 0, stream>>>(Wk, Wtk, D_, U_);
    transpose_cast_f32_bf16<<<dim3(D_ / 64, U_ / 64), blk, 0, stream>>>(Wv, Wtv, D_, U_);

    qkv_gemm<<<dim3((B_ * S_) / MT, U_ / NT, 3), blk, 0, stream>>>(
        Xb, Wtq, Wtk, Wtv, Qb, Kb, Vb);

    transpose_bf16<<<dim3(S_ / 64, U_ / 64, B_), blk, 0, stream>>>(Vb, Vtb, S_, U_);

    const float scale = 1.0f / sqrtf((float)U_);
    scores_gemm<<<dim3(S_ / MT, S_ / NT, B_), blk, 0, stream>>>(Qb, Kb, attn, scale);

    softmax_rows<<<dim3(B_ * S_), blk, 0, stream>>>(attn, Abf);

    context_gemm<<<dim3(S_ / MT, U_ / NT, B_), blk, 0, stream>>>(Abf, Vtb, ctx);
}
